// CrossSSM_80358838108675
// MI455X (gfx1250) — matrix operand order repaired and checked by reading, not compiled, not run on hardware
//
#include <hip/hip_runtime.h>
#include <hip/hip_bf16.h>

// ---------------- problem constants (from setup_inputs) ----------------
#define DIM   96
#define Hh    48
#define Ww    48
#define BATCH 2
#define L     2304          // H*W
#define E     192           // d_inner
#define TWO_E 384
#define Nst   16            // SSM state
#define Rr    6             // dt_rank
#define RP    48            // padded dbc width (R + 2N = 38 -> 48)
#define KB    8             // 4 directions * 2 batch
#define KC    4             // conv taps

typedef _Float16 v16h __attribute__((ext_vector_type(16)));
typedef float    v8f  __attribute__((ext_vector_type(8)));

// ---------------- WMMA fragment loaders (CDNA5 16x16x32 f16 layout) ----
// A: 16x32 (MxK), row-major source. lane&15 = M; kb=(lane>>4)*8;
// VGPR0..3 -> K = kb+0..7 ; VGPR4..7 -> K = 16+kb+0..7 (pairs per VGPR).
__device__ inline v16h load_a_frag(const _Float16* __restrict__ A, int ld, int lane) {
  v16h f;
  int m  = lane & 15;
  int kb = (lane >> 4) << 3;
  const _Float16* p = A + (long)m * ld;
#pragma unroll
  for (int e = 0; e < 8; ++e) {
    int kk = (e < 4) ? (kb + 2 * e) : (16 + kb + 2 * (e - 4));
    f[2 * e]     = p[kk];
    f[2 * e + 1] = p[kk + 1];
  }
  return f;
}
// B: 32x16 (KxN) supplied as Bt row-major (N x K). lane&15 = N;
// kb=(lane>>4)*8; elems 0..7 -> K = kb+0..7 ; elems 8..15 -> K = 16+kb+0..7.
__device__ inline v16h load_b_frag(const _Float16* __restrict__ Bt, int ld, int lane) {
  v16h f;
  int n  = lane & 15;
  int kb = (lane >> 4) << 3;
  const _Float16* p = Bt + (long)n * ld + kb;
#pragma unroll
  for (int e = 0; e < 16; ++e) f[e] = p[(e < 8) ? e : (16 + e - 8)];
  return f;
}

// ---------------- generic batched WMMA GEMM: C = A * Bt^T (+Res) -------
// A: [nb][M][K] f16, Bt: [nb/bdiv][N][K] f16, C/Res: [nb][M][N] f32
__global__ void gemm_wmma_f16(const _Float16* __restrict__ A, long sA,
                              const _Float16* __restrict__ Bt, long sB, int bdiv,
                              const float* __restrict__ Res,
                              float* __restrict__ C, long sC,
                              int M, int N, int K, int nBatch) {
  int tilesN = N >> 4, tilesM = M >> 4;
  int perB   = tilesM * tilesN;
  int wid    = blockIdx.x * (blockDim.x >> 5) + (threadIdx.x >> 5);
  if (wid >= perB * nBatch) return;           // wave-uniform
  int batch = wid / perB;
  int t     = wid - batch * perB;
  int tm = t / tilesN, tn = t - tm * tilesN;
  int lane = threadIdx.x & 31;

  const _Float16* Ab = A  + (long)batch * sA + (long)(tm * 16) * K;
  const _Float16* Bb = Bt + (long)(batch / bdiv) * sB + (long)(tn * 16) * K;

  v8f acc = {};
  for (int k0 = 0; k0 < K; k0 += 32) {
    __builtin_prefetch(Ab + k0 + 32, 0, 1);
    v16h a = load_a_frag(Ab + k0, K, lane);
    v16h b = load_b_frag(Bb + k0, K, lane);
    acc = __builtin_amdgcn_wmma_f32_16x16x32_f16(false, a, false, b,
                                                 (short)0, acc, false, false);
  }
  // D layout: n = lane&15, m = v + 8*(lane>>4)
  int n    = tn * 16 + (lane & 15);
  int mrow = tm * 16 + ((lane >> 4) << 3);
  float*       Cb = C + (long)batch * sC;
  const float* Rb = Res ? (Res + (long)batch * sC) : nullptr;
#pragma unroll
  for (int v = 0; v < 8; ++v) {
    float val = acc[v];
    long  o   = (long)(mrow + v) * N + n;
    if (Rb) val += Rb[o];
    Cb[o] = val;
  }
}

// ---------------- cross-scan forward index map -------------------------
__device__ inline void fwd_map(int k, int l, int& R, int& C) {
  if (k == 0)      { int i = l / 48, jj = l % 48; int j = (i & 1) ? 47 - jj : jj; R = 2 * i;     C = 2 * j; }
  else if (k == 1) { int l2 = 2303 - l; int i = l2 / 48, jj = l2 % 48; int j = (i & 1) ? jj : 47 - jj; R = 2 * i + 1; C = 2 * j + 1; }
  else if (k == 2) { int jd = l / 48, ii = l % 48; int i = (jd & 1) ? 47 - ii : ii; R = 2 * i + 1; C = 2 * jd; }
  else             { int l2 = 2303 - l; int jd = l2 / 48, ii = l2 % 48; int i = (jd & 1) ? ii : 47 - ii; R = 2 * i; C = 2 * jd + 1; }
}
__device__ inline void inv_map(int R, int C, int& k, int& l) {
  int pr = R & 1, pc = C & 1;
  int i = R >> 1, j = C >> 1;
  if (pr == 0 && pc == 0) { k = 0; l = i * 48 + ((i & 1) ? 47 - j : j); }
  else if (pr == 1 && pc == 1) { k = 1; l = 2303 - (i * 48 + ((i & 1) ? j : 47 - j)); }
  else if (pr == 1 && pc == 0) { k = 2; l = j * 48 + ((j & 1) ? 47 - i : i); }
  else { k = 3; l = 2303 - (j * 48 + ((j & 1) ? i : 47 - i)); }
}

// ---------------- stage kernels ---------------------------------------
// weight conversion
__global__ void cvt_f16_kernel(const float* __restrict__ s, _Float16* __restrict__ d, int n) {
  int i = blockIdx.x * blockDim.x + threadIdx.x;
  if (i < n) d[i] = (_Float16)s[i];
}
__global__ void wx_pad_kernel(const float* __restrict__ Wx, _Float16* __restrict__ Wxh) {
  int i = blockIdx.x * blockDim.x + threadIdx.x;       // 4*48*192
  if (i >= 4 * RP * E) return;
  int e = i % E, r = (i / E) % RP, k = i / (E * RP);
  Wxh[i] = (r < (Rr + 2 * Nst)) ? (_Float16)Wx[((long)k * (Rr + 2 * Nst) + r) * E + e] : (_Float16)0.0f;
}
__global__ void aneg_kernel(const float* __restrict__ A_log, float* __restrict__ Aneg, int n) {
  int i = blockIdx.x * blockDim.x + threadIdx.x;
  if (i < n) Aneg[i] = -__expf(A_log[i]);
}

// cross-scan gather + rmsnorm: one block per (kb,l), 96 threads over channels
__global__ void prep_kernel(const float* __restrict__ xs, const float* __restrict__ kp,
                            const float* __restrict__ norm_w,
                            float* __restrict__ xb, _Float16* __restrict__ xn) {
  int c  = threadIdx.x;
  int bi = blockIdx.x;
  int l  = bi % L;
  int kb = bi / L;
  int k = kb >> 1, b = kb & 1;
  int R, C; fwd_map(k, l, R, C);
  int r0 = (R >= 48) ? R - 48 : R;
  int c0 = (C >= 48) ? C - 48 : C;
  int sel = ((R >= 48) != (C >= 48)) ? 1 : 0;   // x1 if quadrants differ
  int sb  = sel ? (b + BATCH) : b;
  int pos = r0 * 48 + c0;
  float val = xs[((long)sb * DIM + c) * L + pos] + kp[(long)c * L + pos];
  __shared__ float sd[DIM];
  sd[c] = val * val;
  __syncthreads();
  float sum = 0.f;
#pragma unroll 8
  for (int i = 0; i < DIM; ++i) sum += sd[i];
  float rs = rsqrtf(sum / (float)DIM + 1e-5f);
  long o = ((long)kb * L + l) * DIM + c;
  xb[o] = val;
  xn[o] = (_Float16)(val * rs * norm_w[k * DIM + c]);
}

// depthwise causal conv(4) + bias + SiLU on xz[..., :E]
__global__ void conv_silu_kernel(const float* __restrict__ xz,
                                 const float* __restrict__ cw, const float* __restrict__ cb,
                                 float* __restrict__ xc, _Float16* __restrict__ xinh) {
  long idx = (long)blockIdx.x * blockDim.x + threadIdx.x;  // KB*L*E
  if (idx >= (long)KB * L * E) return;
  int e  = idx % E;
  int l  = (idx / E) % L;
  int kb = idx / ((long)E * L);
  int k  = kb >> 1;
  const float* w = cw + ((long)k * E + e) * KC;
  float s = cb[k * E + e];
#pragma unroll
  for (int t = 0; t < KC; ++t) {
    int li = l - (KC - 1) + t;
    if (li >= 0) s += w[t] * xz[((long)kb * L + li) * TWO_E + e];
  }
  float a = s / (1.f + __expf(-s));   // SiLU
  xc[idx]   = a;
  xinh[idx] = (_Float16)a;
}

// dt = softplus(dbc[:, :R] @ Wdt^T + bdt)
__global__ void dt_kernel(const float* __restrict__ dbc, const float* __restrict__ Wdt,
                          const float* __restrict__ bdt, float* __restrict__ dt) {
  long idx = (long)blockIdx.x * blockDim.x + threadIdx.x;  // KB*L*E
  if (idx >= (long)KB * L * E) return;
  int e  = idx % E;
  int l  = (idx / E) % L;
  int kb = idx / ((long)E * L);
  int k  = kb >> 1;
  const float* d = dbc + ((long)kb * L + l) * RP;
  const float* w = Wdt + ((long)k * E + e) * Rr;
  float s = bdt[k * E + e];
#pragma unroll
  for (int r = 0; r < Rr; ++r) s += d[r] * w[r];
  dt[idx] = (s > 20.f) ? s : log1pf(__expf(s));
}

// sequential selective scan: 8 blocks (kb), 192 lanes (channel e), 16 states in regs
__global__ void scan_kernel(const float* __restrict__ dt, const float* __restrict__ xc,
                            const float* __restrict__ dbc, const float* __restrict__ xz,
                            const float* __restrict__ Aneg, const float* __restrict__ Dp,
                            _Float16* __restrict__ yact) {
  int kb = blockIdx.x;
  int e  = threadIdx.x;
  int k  = kb >> 1;
  float an[Nst];
#pragma unroll
  for (int n = 0; n < Nst; ++n) an[n] = Aneg[((long)k * E + e) * Nst + n];
  float dp = Dp[k * E + e];
  float h[Nst];
#pragma unroll
  for (int n = 0; n < Nst; ++n) h[n] = 0.f;

  const float* dtp = dt  + (long)kb * L * E;
  const float* xp  = xc  + (long)kb * L * E;
  const float* dbp = dbc + (long)kb * L * RP;
  const float* zp  = xz  + (long)kb * L * TWO_E + E;
  _Float16*    yp  = yact + (long)kb * L * E;

  for (int l = 0; l < L; ++l) {
    float dtv = dtp[(long)l * E + e];
    float xv  = xp[(long)l * E + e];
    float dtx = dtv * xv;
    const float* bb = dbp + (long)l * RP + Rr;
    const float* cc = dbp + (long)l * RP + Rr + Nst;
    float acc = 0.f;
#pragma unroll
    for (int n = 0; n < Nst; ++n) {
      h[n] = __expf(dtv * an[n]) * h[n] + dtx * bb[n];
      acc += h[n] * cc[n];
    }
    float y = acc + dp * xv;
    float z = zp[(long)l * TWO_E + e];
    float sg = z / (1.f + __expf(-z));   // SiLU(z)
    yp[(long)l * E + e] = (_Float16)(y * sg);
  }
}

// cross-merge: gather two composite positions per output pixel, sum
__global__ void merge_kernel(const float* __restrict__ ybr, _Float16* __restrict__ ymg) {
  long idx = (long)blockIdx.x * blockDim.x + threadIdx.x;   // 4*L*DIM
  if (idx >= (long)4 * L * DIM) return;
  int c  = idx % DIM;
  int p  = (idx / DIM) % L;
  int bb = idx / ((long)DIM * L);     // bb = tgt*2 + b  (yc = concat([y0,y1],0))
  int b = bb & 1, tgt = bb >> 1;
  int r0 = p / 48, c0 = p % 48;
  int R1 = r0,      C1 = tgt ? c0 + 48 : c0;
  int R2 = r0 + 48, C2 = tgt ? c0 : c0 + 48;
  int k1, l1, k2, l2;
  inv_map(R1, C1, k1, l1);
  inv_map(R2, C2, k2, l2);
  float v = ybr[(((long)(k1 * 2 + b)) * L + l1) * DIM + c] +
            ybr[(((long)(k2 * 2 + b)) * L + l2) * DIM + c];
  ymg[((long)bb * L + p) * DIM + c] = (_Float16)v;
}

// final dual-GEMM gate: out = (yc@Gw1^T + b1) * sigmoid(yc@Gw2^T + b2)
__global__ void gate_kernel(const _Float16* __restrict__ ymg,
                            const _Float16* __restrict__ G1, const _Float16* __restrict__ G2,
                            const float* __restrict__ Gb1, const float* __restrict__ Gb2,
                            float* __restrict__ out) {
  const int tilesN = DIM / 16;        // 6
  const int tilesM = (4 * L) / 16;    // 576
  int wid = blockIdx.x * (blockDim.x >> 5) + (threadIdx.x >> 5);
  if (wid >= tilesM * tilesN) return;
  int tm = wid / tilesN, tn = wid - tm * tilesN;
  int lane = threadIdx.x & 31;

  const _Float16* Ab = ymg + (long)(tm * 16) * DIM;
  const _Float16* B1 = G1  + (long)(tn * 16) * DIM;
  const _Float16* B2 = G2  + (long)(tn * 16) * DIM;
  v8f ag = {}, as = {};
  for (int k0 = 0; k0 < DIM; k0 += 32) {
    v16h a  = load_a_frag(Ab + k0, DIM, lane);
    v16h b1 = load_b_frag(B1 + k0, DIM, lane);
    v16h b2 = load_b_frag(B2 + k0, DIM, lane);
    ag = __builtin_amdgcn_wmma_f32_16x16x32_f16(false, a, false, b1, (short)0, ag, false, false);
    as = __builtin_amdgcn_wmma_f32_16x16x32_f16(false, a, false, b2, (short)0, as, false, false);
  }
  int n    = tn * 16 + (lane & 15);          // output channel d
  int mrow = tm * 16 + ((lane >> 4) << 3);
  float gb1 = Gb1[n], gb2 = Gb2[n];
#pragma unroll
  for (int v = 0; v < 8; ++v) {
    int m  = mrow + v;
    int bb = m / L;
    int p  = m - bb * L;
    float g = ag[v] + gb1;
    float s = as[v] + gb2;
    out[((long)(bb * DIM + n)) * L + p] = g * (1.f / (1.f + __expf(-s)));
  }
}

// ---------------- host driver ------------------------------------------
static inline int gemm_blocks(int M, int N, int nb) {
  int waves = (M / 16) * (N / 16) * nb;
  return (waves + 7) / 8;   // 8 waves per 256-thread block
}

extern "C" void kernel_launch(void* const* d_in, const int* in_sizes, int n_in,
                              void* d_out, int out_size, void* d_ws, size_t ws_size,
                              hipStream_t stream) {
  const float* xs     = (const float*)d_in[0];
  const float* kp     = (const float*)d_in[1];
  const float* norm_w = (const float*)d_in[2];
  const float* Win    = (const float*)d_in[3];
  const float* conv_w = (const float*)d_in[4];
  const float* conv_b = (const float*)d_in[5];
  const float* Wx     = (const float*)d_in[6];
  const float* Wdt    = (const float*)d_in[7];
  const float* bdt    = (const float*)d_in[8];
  const float* A_log  = (const float*)d_in[9];
  const float* Dp     = (const float*)d_in[10];
  const float* Wout   = (const float*)d_in[11];
  const float* Gw1    = (const float*)d_in[12];
  const float* Gb1    = (const float*)d_in[13];
  const float* Gw2    = (const float*)d_in[14];
  const float* Gb2    = (const float*)d_in[15];
  float* out = (float*)d_out;

  // ---- carve workspace ----
  char* p = (char*)d_ws;
  auto carve = [&](size_t bytes) -> void* {
    void* r = (void*)p;
    p += (bytes + 255) & ~(size_t)255;
    return r;
  };
  float* xb   = (float*)carve((size_t)KB * L * DIM  * 4);
  float* xz   = (float*)carve((size_t)KB * L * TWO_E * 4);
  float* xc   = (float*)carve((size_t)KB * L * E    * 4);
  float* dbc  = (float*)carve((size_t)KB * L * RP   * 4);
  float* dt   = (float*)carve((size_t)KB * L * E    * 4);
  float* ybr  = (float*)carve((size_t)KB * L * DIM  * 4);
  float* Aneg = (float*)carve((size_t)4 * E * Nst   * 4);
  _Float16* xn    = (_Float16*)carve((size_t)KB * L * DIM * 2);
  _Float16* xinh  = (_Float16*)carve((size_t)KB * L * E   * 2);
  _Float16* yacth = (_Float16*)carve((size_t)KB * L * E   * 2);
  _Float16* ymg   = (_Float16*)carve((size_t)4 * L * DIM  * 2);
  _Float16* Winh  = (_Float16*)carve((size_t)4 * TWO_E * DIM * 2);
  _Float16* Wxh   = (_Float16*)carve((size_t)4 * RP * E  * 2);
  _Float16* Wouth = (_Float16*)carve((size_t)4 * DIM * E * 2);
  _Float16* G1h   = (_Float16*)carve((size_t)DIM * DIM * 2);
  _Float16* G2h   = (_Float16*)carve((size_t)DIM * DIM * 2);

  // ---- weight prep ----
  int nWin = 4 * TWO_E * DIM, nWout = 4 * DIM * E, nG = DIM * DIM;
  cvt_f16_kernel<<<(nWin + 255) / 256, 256, 0, stream>>>(Win, Winh, nWin);
  cvt_f16_kernel<<<(nWout + 255) / 256, 256, 0, stream>>>(Wout, Wouth, nWout);
  cvt_f16_kernel<<<(nG + 255) / 256, 256, 0, stream>>>(Gw1, G1h, nG);
  cvt_f16_kernel<<<(nG + 255) / 256, 256, 0, stream>>>(Gw2, G2h, nG);
  wx_pad_kernel<<<(4 * RP * E + 255) / 256, 256, 0, stream>>>(Wx, Wxh);
  aneg_kernel<<<(4 * E * Nst + 255) / 256, 256, 0, stream>>>(A_log, Aneg, 4 * E * Nst);

  // ---- stage 1: cross-scan + rmsnorm ----
  prep_kernel<<<KB * L, DIM, 0, stream>>>(xs, kp, norm_w, xb, xn);

  // ---- stage 2: in-projection (96 -> 384) ----
  gemm_wmma_f16<<<gemm_blocks(L, TWO_E, KB), 256, 0, stream>>>(
      xn, (long)L * DIM, Winh, (long)TWO_E * DIM, 2, nullptr,
      xz, (long)L * TWO_E, L, TWO_E, DIM, KB);

  // ---- stage 3: depthwise conv + SiLU ----
  {
    long n = (long)KB * L * E;
    conv_silu_kernel<<<(int)((n + 255) / 256), 256, 0, stream>>>(xz, conv_w, conv_b, xc, xinh);
  }

  // ---- stage 4: x-projection (192 -> 48 padded) ----
  gemm_wmma_f16<<<gemm_blocks(L, RP, KB), 256, 0, stream>>>(
      xinh, (long)L * E, Wxh, (long)RP * E, 2, nullptr,
      dbc, (long)L * RP, L, RP, E, KB);

  // ---- stage 5: dt projection + softplus ----
  {
    long n = (long)KB * L * E;
    dt_kernel<<<(int)((n + 255) / 256), 256, 0, stream>>>(dbc, Wdt, bdt, dt);
  }

  // ---- stage 6: sequential selective scan + z-gate ----
  scan_kernel<<<KB, E, 0, stream>>>(dt, xc, dbc, xz, Aneg, Dp, yacth);

  // ---- stage 7: out-projection (192 -> 96) + residual ----
  gemm_wmma_f16<<<gemm_blocks(L, DIM, KB), 256, 0, stream>>>(
      yacth, (long)L * E, Wouth, (long)DIM * E, 2, xb,
      ybr, (long)L * DIM, L, DIM, E, KB);

  // ---- stage 8: cross-merge ----
  {
    long n = (long)4 * L * DIM;
    merge_kernel<<<(int)((n + 255) / 256), 256, 0, stream>>>(ybr, ymg);
  }

  // ---- stage 9: dual gate GEMM + sigmoid ----
  gate_kernel<<<gemm_blocks(4 * L, DIM, 1), 256, 0, stream>>>(ymg, G1h, G2h, Gb1, Gb2, out);
}
